// CustomMultiheadAttention_62878321214114
// MI455X (gfx1250) — matrix operand order repaired and checked by reading, not compiled, not run on hardware
//
#include <hip/hip_runtime.h>
#include <math.h>
#include <stdint.h>

// MI455X / gfx1250, wave32 only. Matrix ops via V_WMMA_F32_16X16X32_F16.
typedef __attribute__((ext_vector_type(16))) _Float16 v16h;
typedef __attribute__((ext_vector_type(8)))  float    v8f;

#define BATCH 2
#define SEQ   2048
#define EMB   1024
#define NH    16
#define HD    64
#define MTOT  (BATCH * SEQ)   // 4096 rows of activations

// ---------------------------------------------------------------------------
// Fragment builders matching CDNA5 WMMA VGPR layouts (CDNA5 ISA):
//  A (16x32 f16, MxK): lane L -> M = L&15 ; halfs j<8 : K = (L>>4)*8 + j
//                                          halfs j>=8: K = 16 + (L>>4)*8 + (j-8)
//  B (32x16 f16, KxN): lane L -> N = L&15 ; halfs j<8 : K = (L>>4)*8 + j
//                                          halfs j>=8: K = 16 + (L>>4)*8 + (j-8)
//  C/D (16x16 f32):    lane L -> N = L&15 ; vgpr r   : M = r + (L>>4)*8
// ---------------------------------------------------------------------------
static __device__ __forceinline__ v16h frag_a_f16(const _Float16* base, int ld, int lane) {
  v16h a;
  const int m = lane & 15, q = lane >> 4;
  const _Float16* r = base + m * ld;
#pragma unroll
  for (int j = 0; j < 8; ++j) a[j] = r[q * 8 + j];
#pragma unroll
  for (int j = 0; j < 8; ++j) a[j + 8] = r[16 + q * 8 + j];
  return a;
}

// B[k][n] = base[n*ld + k]  (rows are N, contiguous along K)
static __device__ __forceinline__ v16h frag_b_rows(const _Float16* base, int ld, int lane) {
  v16h b;
  const int n = lane & 15, q = lane >> 4;
  const _Float16* r = base + n * ld;
#pragma unroll
  for (int j = 0; j < 8; ++j) b[j] = r[q * 8 + j];
#pragma unroll
  for (int j = 0; j < 8; ++j) b[j + 8] = r[16 + q * 8 + j];
  return b;
}

// Store layouts for the projection GEMM
#define MODE_QKV 0   // f16 [b,h,s,d]   (Q and K)
#define MODE_VT  1   // f16 [b,h,d,s]   (V transposed: contiguous-K B-frags in P@V)
#define MODE_OUT 2   // f32 [M,EMB]     (final output projection)

// ---------------------------------------------------------------------------
// Tiled GEMM: Y[M,N] = X[M,K] @ W[N,K]^T + bias.
// Block = 128 threads (4 waves), tile 64x64, k-step 32.
// Wave w owns rows [w*16, w*16+16) x 64 cols = 4 WMMA accumulators.
// ---------------------------------------------------------------------------
template <typename TX, int MODE>
__global__ __launch_bounds__(128) void proj_gemm_kernel(
    const TX* __restrict__ X, const float* __restrict__ W,
    const float* __restrict__ bias, void* __restrict__ out, int K) {
  __shared__ _Float16 As[64][40];  // +8 half pad per row to spread banks
  __shared__ _Float16 Bs[64][40];

  const int tid = threadIdx.x, lane = tid & 31, wave = tid >> 5;
  const int m0 = blockIdx.x * 64, n0 = blockIdx.y * 64;
  const int row = tid >> 1, seg = tid & 1;  // 2 threads stage each 32-wide tile row

  v8f acc[4] = {};

  for (int k0 = 0; k0 < K; k0 += 32) {
    const TX*    xs = X + (size_t)(m0 + row) * K + k0 + seg * 16;
    const float* ws = W + (size_t)(n0 + row) * K + k0 + seg * 16;
    if (k0 + 32 < K) {             // gfx1250 global_prefetch_b8 for next k-tile
      __builtin_prefetch(xs + 32, 0, 3);
      __builtin_prefetch(ws + 32, 0, 3);
    }
#pragma unroll
    for (int j = 0; j < 16; ++j) As[row][seg * 16 + j] = (_Float16)xs[j];
#pragma unroll
    for (int j = 0; j < 16; ++j) Bs[row][seg * 16 + j] = (_Float16)ws[j];
    __syncthreads();

    const v16h a = frag_a_f16(&As[wave * 16][0], 40, lane);
#pragma unroll
    for (int t = 0; t < 4; ++t) {
      const v16h b = frag_b_rows(&Bs[t * 16][0], 40, lane);
      acc[t] = __builtin_amdgcn_wmma_f32_16x16x32_f16(false, a, false, b,
                                                      (short)0, acc[t], false, false);
    }
    __syncthreads();
  }

  const int q = lane >> 4, nlane = lane & 15;
#pragma unroll
  for (int t = 0; t < 4; ++t) {
    const int n = n0 + t * 16 + nlane;
    const float bb = bias[n];
#pragma unroll
    for (int r = 0; r < 8; ++r) {
      const int m = m0 + wave * 16 + q * 8 + r;
      const float v = acc[t][r] + bb;
      if (MODE == MODE_QKV) {
        const int b = m >> 11, s = m & (SEQ - 1);
        const int h = n >> 6,  d = n & (HD - 1);
        ((_Float16*)out)[(((size_t)(b * NH + h) * SEQ + s) * HD) + d] = (_Float16)v;
      } else if (MODE == MODE_VT) {
        const int b = m >> 11, s = m & (SEQ - 1);
        const int h = n >> 6,  d = n & (HD - 1);
        ((_Float16*)out)[(((size_t)(b * NH + h) * HD + d) * SEQ) + s] = (_Float16)v;
      } else {
        ((float*)out)[(size_t)m * EMB + n] = v;
      }
    }
  }
}

// ---------------------------------------------------------------------------
// Fused attention: per (b, h, 16-query-row strip)
//   scores = Q Kᵀ * scale  (WMMA, f32 strip 16x2048 kept in LDS: 128 KB)
//   softmax over rows in LDS
//   attn weights written with global_store_async_from_lds_b128, overlapped
//   with the P@V WMMA loop (ASYNCcnt drained at kernel end)
//   ctx = P @ V            (WMMA; V pre-transposed so B-frags are 32B reads)
// Dynamic LDS: 16*2048*4 + 16*64*2 + 16*8*4 = 133,632 B (<320 KB/WGP,
// two blocks co-resident per WGP).
// ---------------------------------------------------------------------------
__global__ __launch_bounds__(128) void attn_kernel(
    const _Float16* __restrict__ qh, const _Float16* __restrict__ kh,
    const _Float16* __restrict__ vt, float* __restrict__ attn_g,
    _Float16* __restrict__ ctx) {
  extern __shared__ char smem_raw[];
  float*    P   = (float*)smem_raw;            // [16][SEQ]
  _Float16* Qt  = (_Float16*)(P + 16 * SEQ);   // [16][HD]
  float*    red = (float*)(Qt + 16 * HD);      // [16][8]

  const int tid = threadIdx.x, lane = tid & 31, wave = tid >> 5;
  const int b = blockIdx.z, h = blockIdx.y, qt = blockIdx.x;
  const size_t headbase = (size_t)(b * NH + h) * SEQ;

  {  // stage Q tile (16 x 64 halfs = 2KB) via async global->LDS DMA
    const char* gsrc = (const char*)(qh + (headbase + (size_t)qt * 16) * HD) + tid * 16;
    const uint32_t loff = (uint32_t)(uintptr_t)Qt + (uint32_t)tid * 16u;
    asm volatile("global_load_async_to_lds_b128 %0, %1, off"
                 :: "v"(loff), "v"(gsrc) : "memory");
    asm volatile("s_wait_asynccnt 0x0" ::: "memory");
  }
  __syncthreads();

  const float scale = 0.125f;  // 1/sqrt(64)
  const int q = lane >> 4, nlane = lane & 15;
  const v16h a0 = frag_a_f16(Qt, HD, lane);        // K = 0..31
  const v16h a1 = frag_a_f16(Qt + 32, HD, lane);   // K = 32..63

  // ---- scores: each wave covers every 4th 16-col key tile ----
  for (int j = wave; j < SEQ / 16; j += 4) {
    const _Float16* kb = kh + (headbase + (size_t)j * 16) * HD;
    const v16h b0 = frag_b_rows(kb, HD, lane);
    const v16h b1 = frag_b_rows(kb + 32, HD, lane);
    v8f acc = {};
    acc = __builtin_amdgcn_wmma_f32_16x16x32_f16(false, a0, false, b0, (short)0, acc, false, false);
    acc = __builtin_amdgcn_wmma_f32_16x16x32_f16(false, a1, false, b1, (short)0, acc, false, false);
#pragma unroll
    for (int r = 0; r < 8; ++r)
      P[(q * 8 + r) * SEQ + j * 16 + nlane] = acc[r] * scale;
  }
  __syncthreads();

  // ---- row softmax: 8 threads per row ----
  {
    const int r = tid >> 3, sub = tid & 7;
    float mx = -3.0e38f;
    for (int c = sub; c < SEQ; c += 8) mx = fmaxf(mx, P[r * SEQ + c]);
    red[r * 8 + sub] = mx;
    __syncthreads();
    mx = red[r * 8];
#pragma unroll
    for (int i = 1; i < 8; ++i) mx = fmaxf(mx, red[r * 8 + i]);

    float s = 0.f;
    for (int c = sub; c < SEQ; c += 8) {
      const float e = __expf(P[r * SEQ + c] - mx);
      P[r * SEQ + c] = e;
      s += e;
    }
    __syncthreads();              // all max-reads of red[] done
    red[r * 8 + sub] = s;
    __syncthreads();
    s = 0.f;
#pragma unroll
    for (int i = 0; i < 8; ++i) s += red[r * 8 + i];
    const float inv = 1.0f / s;
    for (int c = sub; c < SEQ; c += 8) P[r * SEQ + c] *= inv;
  }
  __syncthreads();

  // ---- kick off attn-weights writeback: async LDS->global (overlaps P@V) ----
  {
    const uint32_t pbase = (uint32_t)(uintptr_t)P;
    char* dbase = (char*)(attn_g + (headbase + (size_t)qt * 16) * SEQ);
#pragma unroll 4
    for (int it = 0; it < (16 * SEQ * 4) / (128 * 16); ++it) {  // 64 iters
      const uint32_t loff = pbase + (uint32_t)(it * 128 + tid) * 16u;
      char* dst = dbase + (size_t)(it * 128 + tid) * 16u;
      asm volatile("global_store_async_from_lds_b128 %0, %1, off"
                   :: "v"(dst), "v"(loff) : "memory");
    }
  }

  // ---- ctx = P @ V : wave w owns d-columns [w*16, w*16+16) ----
  {
    const _Float16* vbase = vt + ((size_t)(b * NH + h) * HD + wave * 16) * SEQ;
    v8f acc = {};
    for (int kt = 0; kt < SEQ / 32; ++kt) {
      v16h a;
      const float* prow = P + (lane & 15) * SEQ + kt * 32;
#pragma unroll
      for (int jj = 0; jj < 8; ++jj) a[jj] = (_Float16)prow[q * 8 + jj];
#pragma unroll
      for (int jj = 0; jj < 8; ++jj) a[jj + 8] = (_Float16)prow[16 + q * 8 + jj];

      const v16h bf = frag_b_rows(vbase + kt * 32, SEQ, lane);  // two 16B runs per lane
      acc = __builtin_amdgcn_wmma_f32_16x16x32_f16(false, a, false, bf, (short)0, acc, false, false);
    }
#pragma unroll
    for (int r = 0; r < 8; ++r) {
      const int srow = qt * 16 + q * 8 + r;
      ctx[((size_t)(b * SEQ + srow)) * EMB + h * HD + wave * 16 + nlane] = (_Float16)acc[r];
    }
  }

  // drain async writeback before LDS can be re-allocated
  asm volatile("s_wait_asynccnt 0x0" ::: "memory");
}

// ---------------------------------------------------------------------------
extern "C" void kernel_launch(void* const* d_in, const int* in_sizes, int n_in,
                              void* d_out, int out_size, void* d_ws, size_t ws_size,
                              hipStream_t stream) {
  (void)in_sizes; (void)n_in; (void)out_size; (void)ws_size;
  const float* x_q = (const float*)d_in[0];
  const float* x_k = (const float*)d_in[1];
  const float* x_v = (const float*)d_in[2];
  const float* Wq  = (const float*)d_in[3];
  const float* bq  = (const float*)d_in[4];
  const float* Wk  = (const float*)d_in[5];
  const float* bk  = (const float*)d_in[6];
  const float* Wv  = (const float*)d_in[7];
  const float* bv  = (const float*)d_in[8];
  const float* Wo  = (const float*)d_in[9];
  const float* bo  = (const float*)d_in[10];

  float* out  = (float*)d_out;                       // [4096,1024]
  float* attn = out + (size_t)MTOT * EMB;            // [2,16,2048,2048]

  // workspace: Q/K f16 [b,h,s,d], V f16 transposed [b,h,d,s], ctx f16 [b,s,e]
  _Float16* qhp = (_Float16*)d_ws;
  _Float16* khp = qhp + (size_t)MTOT * EMB;
  _Float16* vtp = khp + (size_t)MTOT * EMB;
  _Float16* ctx = vtp + (size_t)MTOT * EMB;

  const dim3 blk(128);
  const dim3 gproj(MTOT / 64, EMB / 64);

  proj_gemm_kernel<float, MODE_QKV><<<gproj, blk, 0, stream>>>(x_q, Wq, bq, qhp, EMB);
  proj_gemm_kernel<float, MODE_QKV><<<gproj, blk, 0, stream>>>(x_k, Wk, bk, khp, EMB);
  proj_gemm_kernel<float, MODE_VT><<<gproj, blk, 0, stream>>>(x_v, Wv, bv, vtp, EMB);

  const size_t smem = (size_t)16 * SEQ * sizeof(float) + 16 * HD * sizeof(_Float16) +
                      16 * 8 * sizeof(float);
  (void)hipFuncSetAttribute((const void*)attn_kernel,
                            hipFuncAttributeMaxDynamicSharedMemorySize, (int)smem);
  attn_kernel<<<dim3(SEQ / 16, NH, BATCH), blk, smem, stream>>>(qhp, khp, vtp, attn, ctx);

  proj_gemm_kernel<_Float16, MODE_OUT><<<gproj, blk, 0, stream>>>(ctx, Wo, bo, out, EMB);
}
